// GuidedDiffusionNetwork_82446192214391
// MI455X (gfx1250) — compile-verified, matrix operand order checked by reading, not run on hardware
//
#include <hip/hip_runtime.h>
#include <cstddef>

// ---------------- problem constants ----------------
#define BATCH   512
#define NOBJ    16
#define NNODES  (BATCH * NOBJ)      // 8192
#define D1      256
#define D2      128
#define CONDD   512
#define R_REL   26
#define NBASIS  8
#define HIDD    256
#define EMB     256
#define NHEAD   8
#define DHEAD   (EMB / NHEAD)       // 32
#define EPN     (NOBJ * (NOBJ - 1)) // 240 edges per scene
#define NCHUNK  512                 // RGCN node chunk

// ---------------- GEMM tiling ----------------
#define BM 128
#define BN 128
#define BK 32
// 256 threads = 8 waves (wave32); wave grid 2x4; each wave: 4x2 tiles of 16x16 (64x32 region)

typedef __attribute__((ext_vector_type(16))) __bf16 bf16x16;
typedef __attribute__((ext_vector_type(8)))  __bf16 bf16x8;
typedef __attribute__((ext_vector_type(8)))  float  f32x8;

enum { EP_STORE = 0, EP_RELU = 1, EP_ACC = 2, EP_TANH = 3, EP_TANH_RES = 4 };

// ---------- CDNA5 async global->LDS copy (with sync fallback) ----------
#if defined(__has_builtin)
#if __has_builtin(__builtin_amdgcn_global_load_async_to_lds_b128)
#define HAVE_ASYNC_LDS 1
#endif
#endif

__device__ __forceinline__ void cp16(const __bf16* g, __bf16* l) {
#ifdef HAVE_ASYNC_LDS
    typedef int v4i __attribute__((vector_size(16)));
    typedef __attribute__((address_space(1))) v4i gv4i;
    typedef __attribute__((address_space(3))) v4i lv4i;
    __builtin_amdgcn_global_load_async_to_lds_b128(
        (gv4i*)(void*)const_cast<__bf16*>(g),
        (lv4i*)(void*)l, 0, 0);
#else
    *(bf16x8*)l = *(const bf16x8*)g;
#endif
}

__device__ __forceinline__ void wait_async_le4() {
#ifdef HAVE_ASYNC_LDS
#if __has_builtin(__builtin_amdgcn_s_wait_asynccnt)
    __builtin_amdgcn_s_wait_asynccnt(4);
#else
    asm volatile("s_wait_asynccnt 4" ::: "memory");
#endif
#endif
}
__device__ __forceinline__ void wait_async_le0() {
#ifdef HAVE_ASYNC_LDS
#if __has_builtin(__builtin_amdgcn_s_wait_asynccnt)
    __builtin_amdgcn_s_wait_asynccnt(0);
#else
    asm volatile("s_wait_asynccnt 0" ::: "memory");
#endif
#endif
}

// ---------- LDS fragment load: 16 bf16 per lane as two b128 reads ----------
__device__ __forceinline__ bf16x16 ld_frag(const __bf16* p) {
    bf16x8 lo = *(const bf16x8*)p;
    bf16x8 hi = *(const bf16x8*)(p + 16);
    bf16x16 r;
#pragma unroll
    for (int j = 0; j < 8; ++j) { r[j] = lo[j]; r[j + 8] = hi[j]; }
    return r;
}

// ---------------- fused input prep: x + time_embed + maxpool(cond) ----------------
__global__ void make_xin(const float* __restrict__ x, const float* __restrict__ t,
                         const float* __restrict__ cond, float* __restrict__ xin, int d) {
    int idx = blockIdx.x * blockDim.x + threadIdx.x;
    int tot = NNODES * d;
    if (idx >= tot) return;
    int bn = idx / d;
    int c  = idx - bn * d;
    int b  = bn >> 4;
    int half = d >> 1;
    float tb = t[b];
    int k = (c < half) ? c : (c - half);
    float f = __expf(-__logf(10000.0f) * (float)k / (float)(half - 1));
    float te = (c < half) ? __sinf(tb * f) : __cosf(tb * f);
    int k0 = CONDD / d;
    const float* cp = cond + (size_t)bn * CONDD + (size_t)c * k0;
    float mx = cp[0];
    for (int kk = 1; kk < k0; ++kk) mx = fmaxf(mx, cp[kk]);
    xin[idx] = x[idx] + te + mx;
}

// ---------------- RGCN mean aggregation -> bf16 A-rows of folded GEMM ----------------
__global__ void rgcn_agg(const float* __restrict__ X, int dx,
                         const int* __restrict__ esrc, const int* __restrict__ erel,
                         __bf16* __restrict__ Abuf, int chunkStart) {
    extern __shared__ float lds[];             // R*dx sums + R counts
    float* cnt = lds + R_REL * dx;
    const int node = chunkStart + blockIdx.x;
    const int b = node >> 4;
    const int j = node & 15;
    const int tid = threadIdx.x;               // 0..dx-1

    for (int q = tid; q < R_REL * dx; q += dx) lds[q] = 0.0f;
    if (tid < R_REL) cnt[tid] = 0.0f;
    __syncthreads();

    for (int i = 0; i < NOBJ; ++i) {
        if (i == j) continue;
        int e  = b * EPN + i * 15 + ((j > i) ? (j - 1) : j);
        int s  = esrc[e];
        int rr = erel[e];
        lds[rr * dx + tid] += X[(size_t)s * dx + tid];
        if (tid == 0) cnt[rr] += 1.0f;
    }
    __syncthreads();

    __bf16* arow = Abuf + (size_t)blockIdx.x * ((R_REL + 1) * dx);
    for (int r = 0; r < R_REL; ++r)
        arow[r * dx + tid] = (__bf16)(lds[r * dx + tid] * (1.0f / fmaxf(cnt[r], 1.0f)));
    arow[R_REL * dx + tid] = (__bf16)X[(size_t)node * dx + tid];
}

// ------- stacked RGCN weight, transposed bf16: Wt[dout][(R+1)*din] -------
__global__ void build_wext(const float* __restrict__ basis, const float* __restrict__ comp,
                           const float* __restrict__ root, __bf16* __restrict__ Wt,
                           int din, int dout) {
    int idx = blockIdx.x * blockDim.x + threadIdx.x;
    int Ktot = (R_REL + 1) * din;
    int tot = Ktot * dout;
    if (idx >= tot) return;
    int o  = idx / Ktot;           // output channel (row of Wt)
    int kk = idx - o * Ktot;       // k index
    int seg = kk / din;
    int i   = kk - seg * din;
    float acc;
    if (seg < R_REL) {
        acc = 0.0f;
#pragma unroll
        for (int bb = 0; bb < NBASIS; ++bb)
            acc += comp[seg * NBASIS + bb] * basis[((size_t)bb * din + i) * dout + o];
    } else {
        acc = root[(size_t)i * dout + o];
    }
    Wt[idx] = (__bf16)acc;
}

// ---------------- bf16 packing helpers ----------------
__global__ void pack_bf16(const float* __restrict__ s, __bf16* __restrict__ d, int n) {
    int i = blockIdx.x * blockDim.x + threadIdx.x;
    if (i < n) d[i] = (__bf16)s[i];
}
// W [K,N] f32 -> Wt [N,K] bf16
__global__ void pack_bf16_t(const float* __restrict__ W, __bf16* __restrict__ Wt, int K, int N) {
    int i = blockIdx.x * blockDim.x + threadIdx.x;
    if (i >= K * N) return;
    int n = i / K, k = i - n * K;
    Wt[i] = (__bf16)W[(size_t)k * N + n];
}

// ---------------- tiled WMMA GEMM: out[M,Nc] = A[M,K] * Bt[Nc,K]^T ----------------
// A, Bt bf16; double-buffered async LDS staging; 8 v_wmma per wave per K-step.
__global__ __launch_bounds__(256)
void gemm_bf16_tiled(const __bf16* __restrict__ A, const __bf16* __restrict__ Bt,
                     const float* __restrict__ bias, const float* __restrict__ res,
                     float* __restrict__ out, __bf16* __restrict__ dup,
                     int M, int K, int Nc, int mode) {
    __shared__ alignas(16) __bf16 As[2][BM][BK];
    __shared__ alignas(16) __bf16 Bs[2][BN][BK];

    const int tid  = threadIdx.x;
    const int lane = tid & 31;
    const int wid  = tid >> 5;
    const int wm   = wid >> 2;                // 0..1
    const int wn   = wid & 3;                 // 0..3
    const int row0 = blockIdx.x * BM;
    const int col0 = blockIdx.y * BN;

    const int r0 = tid >> 2;                  // staging row 0..63
    const int c0 = (tid & 3) * 8;             // 16B chunk within 64B row
    const int KT = K / BK;

    auto stage = [&](int kt, int buf) {
        const int kb = kt * BK;
#pragma unroll
        for (int s = 0; s < 2; ++s) {
            int r = r0 + s * 64;
            cp16(A  + (size_t)(row0 + r) * K + kb + c0, &As[buf][r][c0]);
            cp16(Bt + (size_t)(col0 + r) * K + kb + c0, &Bs[buf][r][c0]);
        }
    };

    f32x8 zero = {0.f, 0.f, 0.f, 0.f, 0.f, 0.f, 0.f, 0.f};
    f32x8 acc[4][2];
#pragma unroll
    for (int mi = 0; mi < 4; ++mi)
#pragma unroll
        for (int ni = 0; ni < 2; ++ni) acc[mi][ni] = zero;

    stage(0, 0);

    const int fr   = lane & 15;               // fragment row/col within 16
    const int koff = (lane >> 4) << 3;        // 0 or 8 (CDNA5 16-bit operand layout)

    for (int kt = 0; kt < KT; ++kt) {
        const int buf = kt & 1;
        if (kt + 1 < KT) { stage(kt + 1, buf ^ 1); wait_async_le4(); }
        else             { wait_async_le0(); }
        __syncthreads();

        bf16x16 af[4], bfr[2];
#pragma unroll
        for (int mi = 0; mi < 4; ++mi)
            af[mi] = ld_frag(&As[buf][wm * 64 + mi * 16 + fr][koff]);
#pragma unroll
        for (int ni = 0; ni < 2; ++ni)
            bfr[ni] = ld_frag(&Bs[buf][wn * 32 + ni * 16 + fr][koff]);

#pragma unroll
        for (int mi = 0; mi < 4; ++mi)
#pragma unroll
            for (int ni = 0; ni < 2; ++ni)
                acc[mi][ni] = __builtin_amdgcn_wmma_f32_16x16x32_bf16(
                    false, af[mi], false, bfr[ni], (short)0, acc[mi][ni], false, false);

        __syncthreads();
    }

    // epilogue: C/D layout -> lane n = lane&15 ; VGPR i -> m = i + 8*(lane>>4)
    const int nl = lane & 15;
    const int m8 = (lane >> 4) << 3;
#pragma unroll
    for (int mi = 0; mi < 4; ++mi) {
#pragma unroll
        for (int ni = 0; ni < 2; ++ni) {
            int mb = row0 + wm * 64 + mi * 16 + m8;
            int n  = col0 + wn * 32 + ni * 16 + nl;
            float bv = bias ? bias[n] : 0.0f;
#pragma unroll
            for (int i = 0; i < 8; ++i) {
                size_t o = (size_t)(mb + i) * Nc + n;
                float v = acc[mi][ni][i] + bv;
                if (mode == EP_RELU)          v = fmaxf(v, 0.0f);
                else if (mode == EP_ACC)      v += out[o];
                else if (mode == EP_TANH)     v = tanhf(v);
                else if (mode == EP_TANH_RES) v = tanhf(v) + res[o];
                out[o] = v;
                if (dup) dup[o] = (__bf16)v;
            }
        }
    }
}

// ---------------- per-(scene, head) softmax attention core ----------------
__global__ void attn_core(const float* __restrict__ Q, const float* __restrict__ Kv,
                          const float* __restrict__ V, __bf16* __restrict__ O) {
    __shared__ float qs[NOBJ][DHEAD], ks[NOBJ][DHEAD], vs[NOBJ][DHEAD], ss[NOBJ][NOBJ];
    const int b = blockIdx.x / NHEAD;
    const int h = blockIdx.x - b * NHEAD;
    const int tid = threadIdx.x;

    for (int idx = tid; idx < NOBJ * DHEAD; idx += 256) {
        int n = idx >> 5, c = idx & 31;
        size_t g = ((size_t)(b * NOBJ + n)) * EMB + h * DHEAD + c;
        qs[n][c] = Q[g]; ks[n][c] = Kv[g]; vs[n][c] = V[g];
    }
    __syncthreads();

    {
        int qi = tid >> 4, kj = tid & 15;
        float s = 0.0f;
#pragma unroll
        for (int c = 0; c < DHEAD; ++c) s += qs[qi][c] * ks[kj][c];
        ss[qi][kj] = s * 0.17677669529663687f;   // 1/sqrt(32)
    }
    __syncthreads();

    if (tid < NOBJ) {
        float mx = ss[tid][0];
        for (int jj = 1; jj < NOBJ; ++jj) mx = fmaxf(mx, ss[tid][jj]);
        float sum = 0.0f;
        for (int jj = 0; jj < NOBJ; ++jj) { float e = __expf(ss[tid][jj] - mx); ss[tid][jj] = e; sum += e; }
        float inv = 1.0f / sum;
        for (int jj = 0; jj < NOBJ; ++jj) ss[tid][jj] *= inv;
    }
    __syncthreads();

    for (int idx = tid; idx < NOBJ * DHEAD; idx += 256) {
        int n = idx >> 5, c = idx & 31;
        float o = 0.0f;
#pragma unroll
        for (int jj = 0; jj < NOBJ; ++jj) o += ss[n][jj] * vs[jj][c];
        O[((size_t)(b * NOBJ + n)) * EMB + h * DHEAD + c] = (__bf16)o;
    }
}

// ---------------- host-side orchestration ----------------
struct BlockPtrs {
    const float *c1_basis, *c1_bias, *c1_comp, *c1_root;
    const float *c2_basis, *c2_bias, *c2_comp, *c2_root;
    const float *cr_Wk, *cr_Wo, *cr_Wq, *cr_Wv, *cr_bk, *cr_bo, *cr_bq, *cr_bv;
    const float *sf_Wk, *sf_Wo, *sf_Wq, *sf_Wv, *sf_bk, *sf_bo, *sf_bq, *sf_bv;
};

static inline void launch_gemm(const __bf16* A, const __bf16* Bt, const float* bias,
                               const float* res, float* out, __bf16* dup,
                               int M, int K, int Nc, int mode, hipStream_t s) {
    dim3 grid(M / BM, Nc / BN);
    gemm_bf16_tiled<<<grid, 256, 0, s>>>(A, Bt, bias, res, out, dup, M, K, Nc, mode);
}

extern "C" void kernel_launch(void* const* d_in, const int* in_sizes, int n_in,
                              void* d_out, int out_size, void* d_ws, size_t ws_size,
                              hipStream_t stream) {
    (void)in_sizes; (void)n_in; (void)out_size; (void)ws_size;

    const float* xI   = (const float*)d_in[0];
    const float* tI   = (const float*)d_in[1];
    const float* cond = (const float*)d_in[2];
    const int*   esrc = (const int*)d_in[3];
    const int*   erel = (const int*)d_in[4];

    int pi = 5;
    auto nxt = [&]() { return (const float*)d_in[pi++]; };
    BlockPtrs BP[3];
    for (int bi = 0; bi < 3; ++bi) {
        BlockPtrs& p = BP[bi];
        p.c1_basis = nxt(); p.c1_bias = nxt(); p.c1_comp = nxt(); p.c1_root = nxt();
        p.c2_basis = nxt(); p.c2_bias = nxt(); p.c2_comp = nxt(); p.c2_root = nxt();
        p.cr_Wk = nxt(); p.cr_Wo = nxt(); p.cr_Wq = nxt(); p.cr_Wv = nxt();
        p.cr_bk = nxt(); p.cr_bo = nxt(); p.cr_bq = nxt(); p.cr_bv = nxt();
        p.sf_Wk = nxt(); p.sf_Wo = nxt(); p.sf_Wq = nxt(); p.sf_Wv = nxt();
        p.sf_bk = nxt(); p.sf_bo = nxt(); p.sf_bq = nxt(); p.sf_bv = nxt();
    }
    const float* lin1W = nxt(); const float* lin1b = nxt();
    const float* lin2W = nxt(); const float* lin2b = nxt();
    const float* lin3W = nxt(); const float* lin3b = nxt();

    // workspace bump allocator
    char* wp = (char*)d_ws;
    auto alloc = [&](size_t bytes) -> void* {
        void* p = (void*)wp;
        wp += (bytes + 255) & ~(size_t)255;
        return p;
    };
    const size_t NF = (size_t)NNODES;
    float*  xin   = (float*) alloc(NF * 256 * 4);
    __bf16* Abuf  = (__bf16*)alloc((size_t)NCHUNK * ((R_REL + 1) * 256) * 2);
    __bf16* Wext  = (__bf16*)alloc((size_t)((R_REL + 1) * 256) * 256 * 2);
    float*  hbuf  = (float*) alloc(NF * 256 * 4);
    float*  h2    = (float*) alloc(NF * 256 * 4);
    __bf16* h2b   = (__bf16*)alloc(NF * 256 * 2);
    float*  qb    = (float*) alloc(NF * EMB * 4);
    float*  kb    = (float*) alloc(NF * EMB * 4);
    float*  vb    = (float*) alloc(NF * EMB * 4);
    __bf16* abb   = (__bf16*)alloc(NF * EMB * 2);
    float*  x1    = (float*) alloc(NF * 256 * 4);
    __bf16* x1b   = (__bf16*)alloc(NF * 256 * 2);
    float*  x2    = (float*) alloc(NF * 128 * 4);
    float*  xb2   = (float*) alloc(NF * 128 * 4);
    __bf16* xb2b  = (__bf16*)alloc(NF * 128 * 2);
    float*  x4    = (float*) alloc(NF * 256 * 4);
    float*  xb3   = (float*) alloc(NF * 256 * 4);
    __bf16* xb3b  = (__bf16*)alloc(NF * 256 * 2);
    __bf16* condbf= (__bf16*)alloc(NF * CONDD * 2);
    __bf16* wqt   = (__bf16*)alloc((size_t)CONDD * EMB * 2);   // transposed weight slots
    __bf16* wkt   = (__bf16*)alloc((size_t)CONDD * EMB * 2);
    __bf16* wvt   = (__bf16*)alloc((size_t)CONDD * EMB * 2);
    __bf16* wot   = (__bf16*)alloc((size_t)EMB * 256 * 2);

    auto packT = [&](const float* W, __bf16* Wt, int K, int N) {
        pack_bf16_t<<<(K * N + 255) / 256, 256, 0, stream>>>(W, Wt, K, N);
    };

    auto run_block = [&](int d, const float* xcur, const BlockPtrs& p,
                         float* xout, __bf16* xoutb) {
        int tot = NNODES * d;
        make_xin<<<(tot + 255) / 256, 256, 0, stream>>>(xcur, tI, cond, xin, d);

        // conv1: [8192,(R+1)*d] x [(R+1)*d, HID], relu
        int K1 = (R_REL + 1) * d;
        build_wext<<<(K1 * HIDD + 255) / 256, 256, 0, stream>>>(p.c1_basis, p.c1_comp, p.c1_root, Wext, d, HIDD);
        for (int cs = 0; cs < NNODES; cs += NCHUNK) {
            rgcn_agg<<<NCHUNK, d, (R_REL * d + R_REL) * sizeof(float), stream>>>(xin, d, esrc, erel, Abuf, cs);
            launch_gemm(Abuf, Wext, p.c1_bias, nullptr, hbuf + (size_t)cs * HIDD, nullptr,
                        NCHUNK, K1, HIDD, EP_RELU, stream);
        }

        // conv2: [8192,(R+1)*HID] x [(R+1)*HID, d]
        int K2 = (R_REL + 1) * HIDD;
        build_wext<<<(K2 * d + 255) / 256, 256, 0, stream>>>(p.c2_basis, p.c2_comp, p.c2_root, Wext, HIDD, d);
        for (int cs = 0; cs < NNODES; cs += NCHUNK) {
            rgcn_agg<<<NCHUNK, HIDD, (R_REL * HIDD + R_REL) * sizeof(float), stream>>>(hbuf, HIDD, esrc, erel, Abuf, cs);
            launch_gemm(Abuf, Wext, p.c2_bias, nullptr, h2 + (size_t)cs * d, h2b + (size_t)cs * d,
                        NCHUNK, K2, d, EP_STORE, stream);
        }

        // self-attention
        packT(p.sf_Wq, wqt, d, EMB);
        launch_gemm(h2b, wqt, p.sf_bq, nullptr, qb, nullptr, NNODES, d, EMB, EP_STORE, stream);
        packT(p.sf_Wk, wkt, d, EMB);
        launch_gemm(h2b, wkt, p.sf_bk, nullptr, kb, nullptr, NNODES, d, EMB, EP_STORE, stream);
        packT(p.sf_Wv, wvt, d, EMB);
        launch_gemm(h2b, wvt, p.sf_bv, nullptr, vb, nullptr, NNODES, d, EMB, EP_STORE, stream);
        attn_core<<<BATCH * NHEAD, 256, 0, stream>>>(qb, kb, vb, abb);
        packT(p.sf_Wo, wot, EMB, d);
        launch_gemm(abb, wot, p.sf_bo, nullptr, xout, nullptr, NNODES, EMB, d, EP_STORE, stream);

        // cross-attention (kv from obj_cond), accumulate
        packT(p.cr_Wq, wqt, d, EMB);
        launch_gemm(h2b, wqt, p.cr_bq, nullptr, qb, nullptr, NNODES, d, EMB, EP_STORE, stream);
        packT(p.cr_Wk, wkt, CONDD, EMB);
        launch_gemm(condbf, wkt, p.cr_bk, nullptr, kb, nullptr, NNODES, CONDD, EMB, EP_STORE, stream);
        packT(p.cr_Wv, wvt, CONDD, EMB);
        launch_gemm(condbf, wvt, p.cr_bv, nullptr, vb, nullptr, NNODES, CONDD, EMB, EP_STORE, stream);
        attn_core<<<BATCH * NHEAD, 256, 0, stream>>>(qb, kb, vb, abb);
        packT(p.cr_Wo, wot, EMB, d);
        launch_gemm(abb, wot, p.cr_bo, nullptr, xout, xoutb, NNODES, EMB, d, EP_ACC, stream);
    };

    // ---- network ----
    pack_bf16<<<(NNODES * CONDD + 255) / 256, 256, 0, stream>>>(cond, condbf, NNODES * CONDD);

    run_block(D1, xI, BP[0], x1, x1b);
    packT(lin1W, wqt, D1, D2);
    launch_gemm(x1b, wqt, lin1b, nullptr, x2, nullptr, NNODES, D1, D2, EP_TANH, stream);

    run_block(D2, x2, BP[1], xb2, xb2b);
    packT(lin2W, wqt, D2, D1);
    launch_gemm(xb2b, wqt, lin2b, x1, x4, nullptr, NNODES, D2, D1, EP_TANH_RES, stream);

    run_block(D1, x4, BP[2], xb3, xb3b);
    packT(lin3W, wqt, D1, D1);
    launch_gemm(xb3b, wqt, lin3b, nullptr, (float*)d_out, nullptr, NNODES, D1, D1, EP_TANH, stream);
}
